// MultiHeadSelfAttention_25769803796
// MI455X (gfx1250) — compile-verified, matrix operand order checked by reading, not run on hardware
//
#include <hip/hip_runtime.h>

#define D_SEQ   2048
#define D_B     4
#define D_E     1024
#define D_H     16
#define D_HD    64
#define D_ROWS  (D_SEQ * D_B)     /* 8192 */
#define D_3E    (3 * D_E)         /* 3072 */

typedef __bf16 bf16_t;
typedef __attribute__((ext_vector_type(16))) __bf16 v16bf;
typedef __attribute__((ext_vector_type(8)))  __bf16 v8bf;
typedef __attribute__((ext_vector_type(8)))  float  v8f;

// ---------------------------------------------------------------------------
// Load a 16x32 bf16 tile in the CDNA5 WMMA A/B operand layout.
// Source is row-major with leading dimension `ld` (elements); `base` points at
// element [row0][k0].  Lane r (0..15) holds row/col r; the upper half-wave
// (hi=1) holds K = 8..15 / 24..31, the lower half K = 0..7 / 16..23.
// Two 16-byte loads per lane.
// ---------------------------------------------------------------------------
static __device__ __forceinline__ v16bf load_tile(const bf16_t* base, int ld) {
  const int lane = threadIdx.x & 31;
  const int row  = lane & 15;
  const int hi   = lane >> 4;
  const bf16_t* p = base + (size_t)row * ld + hi * 8;
  v8bf lo = *(const v8bf*)(p);        // K = hi*8 + 0..7
  v8bf up = *(const v8bf*)(p + 16);   // K = 16 + hi*8 + 0..7
  v16bf out;
#pragma unroll
  for (int i = 0; i < 8; ++i) { out[i] = lo[i]; out[i + 8] = up[i]; }
  return out;
}

static __device__ __forceinline__ v8f wmma_bf16(v16bf a, v16bf b, v8f c) {
  // D = A(16x32 bf16) * B(32x16 bf16) + C(16x16 f32)
  return __builtin_amdgcn_wmma_f32_16x16x32_bf16(
      /*neg_a=*/false, a, /*neg_b=*/false, b,
      /*c_mod=*/(short)0, c, /*reuse_a=*/false, /*reuse_b=*/false);
}

// ---------------------------------------------------------------------------
// fp32 -> bf16 conversion (grid-stride)
// ---------------------------------------------------------------------------
__global__ void cvt_f32_bf16(const float* __restrict__ in,
                             bf16_t* __restrict__ out, int n) {
  int i = blockIdx.x * blockDim.x + threadIdx.x;
  const int stride = gridDim.x * blockDim.x;
  for (; i < n; i += stride) out[i] = (bf16_t)in[i];
}

// ---------------------------------------------------------------------------
// QKV projection:  qkv[m, n] = X[m, :] . W_in[n, :] + b_in[n]
// M = 8192 (row m = s*B + b), N = 3072, K = 1024.
// Epilogue scatters:  Q (scaled) -> [B,H,S,64] bf16
//                     K          -> [B,H,S,64] bf16
//                     V          -> [B,H,64,S] bf16  (transposed for PV GEMM)
// Wave tile: 32 rows x 64 cols (2x4 accumulators).
// ---------------------------------------------------------------------------
__global__ __launch_bounds__(256) void qkv_gemm(
    const bf16_t* __restrict__ X, const bf16_t* __restrict__ Win,
    const float* __restrict__ bin,
    bf16_t* __restrict__ Q, bf16_t* __restrict__ K, bf16_t* __restrict__ Vt) {
  const int wid = blockIdx.x * 8 + (threadIdx.x >> 5);
  const int TN  = D_3E / 64;                 // 48 col tiles
  const int tm  = wid / TN, tn = wid % TN;
  const int m0  = tm * 32, n0 = tn * 64;

  v8f acc[2][4] = {};
  for (int k0 = 0; k0 < D_E; k0 += 32) {
    v16bf a0 = load_tile(X + (size_t)m0 * D_E + k0, D_E);
    v16bf a1 = load_tile(X + (size_t)(m0 + 16) * D_E + k0, D_E);
#pragma unroll
    for (int ni = 0; ni < 4; ++ni) {
      v16bf b = load_tile(Win + (size_t)(n0 + ni * 16) * D_E + k0, D_E);
      acc[0][ni] = wmma_bf16(a0, b, acc[0][ni]);
      acc[1][ni] = wmma_bf16(a1, b, acc[1][ni]);
    }
  }

  const int lane = threadIdx.x & 31, r = lane & 15, hi = lane >> 4;
  const float scaling = 0.125f;              // 64^-0.5
#pragma unroll
  for (int mi = 0; mi < 2; ++mi) {
#pragma unroll
    for (int ni = 0; ni < 4; ++ni) {
      const int n     = n0 + ni * 16 + r;
      const int which = n >> 10;             // 0=q 1=k 2=v
      const int e     = n & (D_E - 1);
      const int h     = e >> 6, d = e & 63;
      const float bias = bin[n];
#pragma unroll
      for (int v = 0; v < 8; ++v) {
        const int m = m0 + mi * 16 + v + 8 * hi;   // row = s*B + b
        const int s = m >> 2, b = m & 3;
        float val = acc[mi][ni][v] + bias;
        const size_t bh = (size_t)(b * D_H + h);
        if (which == 0) {
          val *= scaling;
          Q[(bh * D_SEQ + s) * D_HD + d] = (bf16_t)val;
        } else if (which == 1) {
          K[(bh * D_SEQ + s) * D_HD + d] = (bf16_t)val;
        } else {
          Vt[(bh * D_HD + d) * D_SEQ + s] = (bf16_t)val;
        }
      }
    }
  }
}

// ---------------------------------------------------------------------------
// Flash attention per (b,h).  grid = (B*H, S/128), block = 256 (8 waves).
// Each wave owns 16 query rows; online softmax over 32-column K tiles.
// Scores/P never touch HBM; P is re-laid-out C->A via 1KB per-wave LDS.
// ---------------------------------------------------------------------------
__global__ __launch_bounds__(256) void flash_attn(
    const bf16_t* __restrict__ Q, const bf16_t* __restrict__ K,
    const bf16_t* __restrict__ Vt, const float* __restrict__ mask,
    const unsigned char* __restrict__ pad, bf16_t* __restrict__ attn) {
  __shared__ bf16_t ldsP[8][16 * 32];        // per-wave P tile (bf16, 16x32)

  const int wave = threadIdx.x >> 5;
  const int lane = threadIdx.x & 31, r = lane & 15, hi = lane >> 4;
  const int bh = blockIdx.x;                 // 0..63
  const int b  = bh >> 4, h = bh & 15;
  const int s0 = blockIdx.y * 128 + wave * 16;

  const bf16_t* Qb = Q  + (size_t)bh * D_SEQ * D_HD;
  const bf16_t* Kb = K  + (size_t)bh * D_SEQ * D_HD;
  const bf16_t* Vb = Vt + (size_t)bh * D_HD * D_SEQ;
  const float*  Mb = mask + (size_t)b * D_SEQ * D_SEQ;
  const unsigned char* Pb = pad + (size_t)b * D_SEQ;

  // Q strip (16 x 64) stays in registers for the whole pass.
  const v16bf qa0 = load_tile(Qb + (size_t)s0 * D_HD,      D_HD);
  const v16bf qa1 = load_tile(Qb + (size_t)s0 * D_HD + 32, D_HD);

  v8f O[4] = {};
  float mrun[8], lrun[8];
#pragma unroll
  for (int v = 0; v < 8; ++v) { mrun[v] = -3.0e38f; lrun[v] = 0.0f; }

  for (int t0 = 0; t0 < D_SEQ; t0 += 32) {
    // ---- scores: S(16x32) = Qstrip(16x64) . K[t0:t0+32, :]^T -------------
    v8f S[2];
#pragma unroll
    for (int nt = 0; nt < 2; ++nt) {
      v16bf kb0 = load_tile(Kb + (size_t)(t0 + nt * 16) * D_HD,      D_HD);
      v16bf kb1 = load_tile(Kb + (size_t)(t0 + nt * 16) * D_HD + 32, D_HD);
      v8f s = {};
      s = wmma_bf16(qa0, kb0, s);
      s = wmma_bf16(qa1, kb1, s);
      S[nt] = s;
    }

    // ---- + attention_mask, padding mask; tile row max ---------------------
    float tmax[8];
#pragma unroll
    for (int v = 0; v < 8; ++v) tmax[v] = -3.0e38f;
#pragma unroll
    for (int nt = 0; nt < 2; ++nt) {
      const int col = t0 + nt * 16 + r;
      const bool padded = Pb[col] != 0;
#pragma unroll
      for (int v = 0; v < 8; ++v) {
        const int row = s0 + v + 8 * hi;
        float sc = S[nt][v] + Mb[(size_t)row * D_SEQ + col];
        sc = padded ? -10000.0f : sc;
        S[nt][v] = sc;
        tmax[v] = fmaxf(tmax[v], sc);
      }
    }
    // reduce across the 16 column-lanes of each half-wave
#pragma unroll
    for (int v = 0; v < 8; ++v) {
#pragma unroll
      for (int xm = 1; xm <= 8; xm <<= 1)
        tmax[v] = fmaxf(tmax[v], __shfl_xor(tmax[v], xm, 32));
    }

    // ---- online softmax update -------------------------------------------
    float alpha[8], rs[8];
#pragma unroll
    for (int v = 0; v < 8; ++v) {
      const float mn = fmaxf(mrun[v], tmax[v]);
      alpha[v] = __expf(mrun[v] - mn);
      mrun[v]  = mn;
      rs[v]    = 0.0f;
    }
#pragma unroll
    for (int nt = 0; nt < 2; ++nt)
#pragma unroll
      for (int v = 0; v < 8; ++v) {
        const float p = __expf(S[nt][v] - mrun[v]);
        S[nt][v] = p;
        rs[v] += p;
      }
#pragma unroll
    for (int v = 0; v < 8; ++v) {
#pragma unroll
      for (int xm = 1; xm <= 8; xm <<= 1)
        rs[v] += __shfl_xor(rs[v], xm, 32);
      lrun[v] = lrun[v] * alpha[v] + rs[v];
    }
#pragma unroll
    for (int nt2 = 0; nt2 < 4; ++nt2)
#pragma unroll
      for (int v = 0; v < 8; ++v) O[nt2][v] *= alpha[v];

    // ---- P: C-layout f32 -> bf16 row-major in LDS -> A-layout reload -----
    bf16_t* lp = &ldsP[wave][0];
#pragma unroll
    for (int nt = 0; nt < 2; ++nt)
#pragma unroll
      for (int v = 0; v < 8; ++v)
        lp[(v + 8 * hi) * 32 + nt * 16 + r] = (bf16_t)S[nt][v];
    const v16bf pa = load_tile(lp, 32);   // DS ops are in-order within a wave

    // ---- O(16x64) += P(16x32) . V[t0:t0+32, :]  (B tiles from Vt rows) ---
#pragma unroll
    for (int nt2 = 0; nt2 < 4; ++nt2) {
      v16bf vb = load_tile(Vb + (size_t)(nt2 * 16) * D_SEQ + t0, D_SEQ);
      O[nt2] = wmma_bf16(pa, vb, O[nt2]);
    }
  }

  // ---- normalize and store to attn[S, B, E] as bf16 -----------------------
#pragma unroll
  for (int nt2 = 0; nt2 < 4; ++nt2) {
#pragma unroll
    for (int v = 0; v < 8; ++v) {
      const int s = s0 + v + 8 * hi;
      const int e = h * D_HD + nt2 * 16 + r;
      const float val = O[nt2][v] / lrun[v];
      attn[((size_t)s * D_B + b) * D_E + e] = (bf16_t)val;
    }
  }
}

// ---------------------------------------------------------------------------
// Out projection: out[m, n] = attn[m, :] . W_out[n, :] + b_out[n]  (fp32 out)
// ---------------------------------------------------------------------------
__global__ __launch_bounds__(256) void out_gemm(
    const bf16_t* __restrict__ A, const bf16_t* __restrict__ Wout,
    const float* __restrict__ bout, float* __restrict__ out) {
  const int wid = blockIdx.x * 8 + (threadIdx.x >> 5);
  const int TN  = D_E / 64;                  // 16 col tiles
  const int tm  = wid / TN, tn = wid % TN;
  const int m0  = tm * 32, n0 = tn * 64;

  v8f acc[2][4] = {};
  for (int k0 = 0; k0 < D_E; k0 += 32) {
    v16bf a0 = load_tile(A + (size_t)m0 * D_E + k0, D_E);
    v16bf a1 = load_tile(A + (size_t)(m0 + 16) * D_E + k0, D_E);
#pragma unroll
    for (int ni = 0; ni < 4; ++ni) {
      v16bf b = load_tile(Wout + (size_t)(n0 + ni * 16) * D_E + k0, D_E);
      acc[0][ni] = wmma_bf16(a0, b, acc[0][ni]);
      acc[1][ni] = wmma_bf16(a1, b, acc[1][ni]);
    }
  }

  const int lane = threadIdx.x & 31, r = lane & 15, hi = lane >> 4;
#pragma unroll
  for (int mi = 0; mi < 2; ++mi) {
#pragma unroll
    for (int ni = 0; ni < 4; ++ni) {
      const int n = n0 + ni * 16 + r;
      const float bias = bout[n];
#pragma unroll
      for (int v = 0; v < 8; ++v) {
        const int m = m0 + mi * 16 + v + 8 * hi;
        out[(size_t)m * D_E + n] = acc[mi][ni][v] + bias;
      }
    }
  }
}

// ---------------------------------------------------------------------------
extern "C" void kernel_launch(void* const* d_in, const int* in_sizes, int n_in,
                              void* d_out, int out_size, void* d_ws, size_t ws_size,
                              hipStream_t stream) {
  (void)in_sizes; (void)n_in; (void)out_size; (void)ws_size;
  const float*         query = (const float*)d_in[0];
  const float*         amask = (const float*)d_in[1];
  const unsigned char* pmask = (const unsigned char*)d_in[2];
  const float*         W_in  = (const float*)d_in[3];
  const float*         b_in  = (const float*)d_in[4];
  const float*         W_out = (const float*)d_in[5];
  const float*         b_out = (const float*)d_in[6];
  float* out = (float*)d_out;

  // workspace carve-up (all bf16 buffers), 256B aligned
  char*  ws  = (char*)d_ws;
  size_t off = 0;
  auto alloc = [&](size_t bytes) {
    char* p = ws + off;
    off += (bytes + 255) & ~(size_t)255;
    return p;
  };
  bf16_t* Xbf    = (bf16_t*)alloc((size_t)D_ROWS * D_E * sizeof(bf16_t));
  bf16_t* Winbf  = (bf16_t*)alloc((size_t)D_3E * D_E * sizeof(bf16_t));
  bf16_t* Woutbf = (bf16_t*)alloc((size_t)D_E * D_E * sizeof(bf16_t));
  bf16_t* Qb     = (bf16_t*)alloc((size_t)D_B * D_H * D_SEQ * D_HD * sizeof(bf16_t));
  bf16_t* Kb     = (bf16_t*)alloc((size_t)D_B * D_H * D_SEQ * D_HD * sizeof(bf16_t));
  bf16_t* Vtb    = (bf16_t*)alloc((size_t)D_B * D_H * D_HD * D_SEQ * sizeof(bf16_t));
  bf16_t* Attn   = (bf16_t*)alloc((size_t)D_ROWS * D_E * sizeof(bf16_t));

  // 1) fp32 -> bf16 conversions
  {
    const int nX = D_ROWS * D_E;
    const int nWi = D_3E * D_E;
    const int nWo = D_E * D_E;
    cvt_f32_bf16<<<(nX  + 255) / 256, 256, 0, stream>>>(query, Xbf,    nX);
    cvt_f32_bf16<<<(nWi + 255) / 256, 256, 0, stream>>>(W_in,  Winbf,  nWi);
    cvt_f32_bf16<<<(nWo + 255) / 256, 256, 0, stream>>>(W_out, Woutbf, nWo);
  }

  // 2) QKV projection: (8192/32)*(3072/64) = 12288 wave tiles / 8 per block
  qkv_gemm<<<12288 / 8, 256, 0, stream>>>(Xbf, Winbf, b_in, Qb, Kb, Vtb);

  // 3) flash attention: grid (B*H, S/128)
  flash_attn<<<dim3(D_B * D_H, D_SEQ / 128), 256, 0, stream>>>(
      Qb, Kb, Vtb, amask, pmask, Attn);

  // 4) out projection: (8192/32)*(1024/64) = 4096 wave tiles / 8 per block
  out_gemm<<<4096 / 8, 256, 0, stream>>>(Attn, Woutbf, b_out, out);
}
